// RGCBlock_22711787061763
// MI455X (gfx1250) — matrix operand order repaired and checked by reading, not compiled, not run on hardware
//
#include <hip/hip_runtime.h>

// ---------------------------------------------------------------- types ----
typedef __bf16 bf16;
typedef bf16  v16bf __attribute__((ext_vector_type(16)));
typedef bf16  v8bf  __attribute__((ext_vector_type(8)));
typedef bf16  v4bf  __attribute__((ext_vector_type(4)));
typedef float v8f   __attribute__((ext_vector_type(8)));

// ------------------------------------------------------------ constants ----
static constexpr int BB = 4, LL = 2048, KN = 20;
static constexpr int E  = BB * LL * KN;   // 163840 edge rows
static constexpr int NN = BB * LL;        // 8192 node rows
static constexpr int LK = LL * KN;        // 40960

// stats arena offsets (floats): each entry holds (sum, sumsq) per channel
static constexpr int ST_H0 = 0, ST_H1 = 128, ST_R0E = 256, ST_H3 = 384, ST_R1E = 512;
static constexpr int ST_X = 640, ST_TRG = 896, ST_EOUT = 1152, ST_NEN = 1408;
static constexpr int ST_E0 = 2176, ST_E1 = 2432, ST_R0N = 2688, ST_E3 = 2944, ST_R1N = 3200;
static constexpr int ST_AGG = 3456, ST_G1 = 3712, ST_R0S = 3968, ST_G3 = 4224, ST_R1S = 4480;

// workspace byte offsets
static constexpr size_t SZ_BIG  = (size_t)E * 128 * 4;          // 83,886,080
static constexpr size_t OFF_BUFA = 0;
static constexpr size_t OFF_BUFB = SZ_BIG;
static constexpr size_t OFF_AGG  = 2 * SZ_BIG;
static constexpr size_t OFF_S0   = OFF_AGG + (size_t)NN * 128 * 4;
static constexpr size_t OFF_S1   = OFF_S0  + (size_t)NN * 128 * 4;
static constexpr size_t OFF_DEG  = OFF_S1  + (size_t)NN * 128 * 4;
static constexpr size_t OFF_ST   = OFF_DEG + (size_t)NN * 4;

// ------------------------------------------------------------- helpers ----
__device__ __forceinline__ bf16 f2bf(float f) {
    unsigned u = __builtin_bit_cast(unsigned, f);
    u += 0x7fffu + ((u >> 16) & 1u);            // round-to-nearest-even
    unsigned short h = (unsigned short)(u >> 16);
    return __builtin_bit_cast(bf16, h);
}

// build a 16-element bf16 fragment from two contiguous 8-element runs
__device__ __forceinline__ v16bf frag_from(const bf16* p, int off2) {
    v8bf lo = *(const v8bf*)p;
    v8bf hi = *(const v8bf*)(p + off2);
    return __builtin_shufflevector(lo, hi, 0,1,2,3,4,5,6,7,8,9,10,11,12,13,14,15);
}

// ------------------------------------------------------ WMMA GEMM kernel ---
// Out[row, 0:NDIM] = prenorm(A)[row, 0:KDIM] @ W^T + bias (+ resid)
// AMODE: 0=direct buffer, 1=concat [x_src | edgevec | x_trg] (K=320),
//        2=concat [x_src | edge_out | x_trg] (K=384)
template <int KDIM, int NDIM, int AMODE>
__global__ void __launch_bounds__(128)
__attribute__((amdgpu_waves_per_eu(2, 8)))
gemm_k(const float* __restrict__ Ain, const float* __restrict__ xin,
       const float* __restrict__ evin, const int* __restrict__ adj,
       const float* __restrict__ W, const float* __restrict__ bias,
       const float* __restrict__ stats, const float* __restrict__ bn_g,
       const float* __restrict__ bn_b, float inv_cnt, int do_relu,
       const float* __restrict__ resid, float* __restrict__ Out)
{
    constexpr int KSC = (KDIM % 128 == 0) ? 128 : 64;   // k super-chunk
    constexpr int NSC = KDIM / KSC;
    constexpr int NT  = NDIM / 16;
    constexpr int NKC = KSC / 32;                       // wmma k-steps / chunk
    constexpr int KV  = KSC / 4;                        // float4 groups per row

    __shared__ bf16  As[64][KSC + 8]   __attribute__((aligned(16)));
    __shared__ bf16  Bs[NDIM][KSC + 8] __attribute__((aligned(16)));
    __shared__ float sS[KDIM];
    __shared__ float sB[KDIM];

    const int rowbase = blockIdx.x * 64;
    const int tid = threadIdx.x;
    const int wl = tid & 31, wv = tid >> 5;
    const int m  = wl & 15,  hb = wl >> 4;

    // per-channel BatchNorm scale/shift from raw (sum, sumsq)
    if (stats) {
        for (int c = tid; c < KDIM; c += 128) {
            float s = stats[2 * c], q = stats[2 * c + 1];
            float mu  = s * inv_cnt;
            float var = fmaxf(q * inv_cnt - mu * mu, 0.f);
            float sc  = bn_g[c] * rsqrtf(var + 1e-5f);
            sS[c] = sc;
            sB[c] = bn_b[c] - mu * sc;
        }
    }

    v8f acc[NT];
    #pragma unroll
    for (int nt = 0; nt < NT; ++nt) acc[nt] = (v8f){0.f,0.f,0.f,0.f,0.f,0.f,0.f,0.f};

    for (int sc = 0; sc < NSC; ++sc) {
        __syncthreads();
        // ---- stage A tile: float4 global loads, BN+relu, packed bf16 store ----
        for (int idx = tid; idx < 64 * KV; idx += 128) {
            int r = idx / KV, c4 = (idx - r * KV) * 4;
            int row = rowbase + r, gcol = sc * KSC + c4;
            float4 v;
            if constexpr (AMODE == 0) {
                v = *(const float4*)&Ain[(size_t)row * KDIM + gcol];
            } else if constexpr (AMODE == 1) {
                if (gcol < 128)      v = *(const float4*)&xin[(size_t)(row / KN) * 128 + gcol];
                else if (gcol < 192) v = *(const float4*)&evin[(size_t)row * 64 + (gcol - 128)];
                else { int trg = (row / LK) * LL + adj[row];
                       v = *(const float4*)&xin[(size_t)trg * 128 + (gcol - 192)]; }
            } else {
                if (gcol < 128)      v = *(const float4*)&xin[(size_t)(row / KN) * 128 + gcol];
                else if (gcol < 256) v = *(const float4*)&evin[(size_t)row * 128 + (gcol - 128)];
                else { int trg = (row / LK) * LL + adj[row];
                       v = *(const float4*)&xin[(size_t)trg * 128 + (gcol - 256)]; }
            }
            float e0 = v.x, e1 = v.y, e2 = v.z, e3 = v.w;
            if (stats) {
                e0 = e0 * sS[gcol + 0] + sB[gcol + 0];
                e1 = e1 * sS[gcol + 1] + sB[gcol + 1];
                e2 = e2 * sS[gcol + 2] + sB[gcol + 2];
                e3 = e3 * sS[gcol + 3] + sB[gcol + 3];
                if (do_relu) {
                    e0 = fmaxf(e0, 0.f); e1 = fmaxf(e1, 0.f);
                    e2 = fmaxf(e2, 0.f); e3 = fmaxf(e3, 0.f);
                }
            }
            v4bf pk = { f2bf(e0), f2bf(e1), f2bf(e2), f2bf(e3) };
            *(v4bf*)&As[r][c4] = pk;
        }
        // ---- stage B tile (weights, bf16) : Bs[n][k] = W[n, k] ----
        for (int idx = tid; idx < NDIM * KV; idx += 128) {
            int n = idx / KV, c4 = (idx - n * KV) * 4;
            float4 v = *(const float4*)&W[(size_t)n * KDIM + sc * KSC + c4];
            v4bf pk = { f2bf(v.x), f2bf(v.y), f2bf(v.z), f2bf(v.w) };
            *(v4bf*)&Bs[n][c4] = pk;
        }
        __syncthreads();

        // ---- WMMA: 16x16x32 bf16, fp32 accumulate ----
        // per k-step: batch ALL fragment ds_loads (A + NT B tiles), then the
        // WMMA chain.  sched_group_barrier(0x100)=DS-read group,
        // (0x8)=MFMA/WMMA group: forces load batching + distinct registers,
        // so waits become partial (s_wait_dscnt N) instead of 0 per WMMA.
        #pragma unroll
        for (int kc = 0; kc < NKC; ++kc) {
            v16bf a = frag_from(&As[wv * 16 + m][kc * 32 + hb * 8], 16);
            v16bf bfr[NT];
            #pragma unroll
            for (int nt = 0; nt < NT; ++nt)
                bfr[nt] = frag_from(&Bs[nt * 16 + m][kc * 32 + hb * 8], 16);
            #pragma unroll
            for (int nt = 0; nt < NT; ++nt)
                acc[nt] = __builtin_amdgcn_wmma_f32_16x16x32_bf16(
                    false, a, false, bfr[nt], (short)0, acc[nt], false, false);
            __builtin_amdgcn_sched_group_barrier(0x100, 2 + 2 * NT, 0); // DS reads
            __builtin_amdgcn_sched_group_barrier(0x008, NT, 0);         // WMMAs
        }
    }

    // ---- epilogue: bias (+ residual), fp32 store ----
    #pragma unroll
    for (int nt = 0; nt < NT; ++nt) {
        int col = nt * 16 + m;
        float bn = bias[col];
        #pragma unroll
        for (int j = 0; j < 8; ++j) {
            int row = rowbase + wv * 16 + hb * 8 + j;
            size_t o = (size_t)row * NDIM + col;
            float v = acc[nt][j] + bn;
            if (resid) v += resid[o];
            Out[o] = v;
        }
    }
}

// ------------------------------------------------------ small kernels ------
__global__ void deg_k(const int* __restrict__ adj, int* __restrict__ deg) {
    int e = blockIdx.x * 256 + threadIdx.x;
    if (e < E) { int b = e / LK; atomicAdd(&deg[b * LL + adj[e]], 1); }
}

__global__ void stats_k(const float* __restrict__ buf, int rows, int cols,
                        float* __restrict__ out) {
    int c = blockIdx.x, t = threadIdx.x;
    float s = 0.f, q = 0.f;
    for (int r = t; r < rows; r += 256) {
        float v = buf[(size_t)r * cols + c]; s += v; q += v * v;
    }
    __shared__ float sh[512];
    sh[t] = s; sh[t + 256] = q; __syncthreads();
    for (int o = 128; o; o >>= 1) {
        if (t < o) { sh[t] += sh[t + o]; sh[t + 256] += sh[t + 256 + o]; }
        __syncthreads();
    }
    if (t == 0) { out[2 * c] = sh[0]; out[2 * c + 1] = sh[256]; }
}

__global__ void statsw_k(const float* __restrict__ buf, const int* __restrict__ deg,
                         int rows, int cols, float* __restrict__ out) {
    int c = blockIdx.x, t = threadIdx.x;
    float s = 0.f, q = 0.f;
    for (int r = t; r < rows; r += 256) {
        float w = (float)deg[r];
        float v = buf[(size_t)r * cols + c]; s += w * v; q += w * v * v;
    }
    __shared__ float sh[512];
    sh[t] = s; sh[t + 256] = q; __syncthreads();
    for (int o = 128; o; o >>= 1) {
        if (t < o) { sh[t] += sh[t + o]; sh[t + 256] += sh[t + 256 + o]; }
        __syncthreads();
    }
    if (t == 0) { out[2 * c] = sh[0]; out[2 * c + 1] = sh[256]; }
}

__global__ void assemble_k(float* __restrict__ st) {   // build 384-ch enc_bn0 stats
    int c = threadIdx.x;                               // 384 threads
    float* nen = st + ST_NEN;
    if (c < 128)      { nen[2*c] = (float)KN * st[ST_X + 2*c];
                        nen[2*c+1] = (float)KN * st[ST_X + 2*c + 1]; }
    else if (c < 256) { int d = c - 128; nen[2*c] = st[ST_EOUT + 2*d];
                        nen[2*c+1] = st[ST_EOUT + 2*d + 1]; }
    else              { int d = c - 256; nen[2*c] = st[ST_TRG + 2*d];
                        nen[2*c+1] = st[ST_TRG + 2*d + 1]; }
}

__device__ __forceinline__ void bn_coef(const float* st, const float* g,
                                        const float* b, int ch, float inv_cnt,
                                        float& sc, float& sh) {
    float mu  = st[2 * ch] * inv_cnt;
    float var = fmaxf(st[2 * ch + 1] * inv_cnt - mu * mu, 0.f);
    sc = g[ch] * rsqrtf(var + 1e-5f);
    sh = b[ch] - mu * sc;
}

__global__ void edgeout_k(const float* __restrict__ ev, const float* __restrict__ r1,
                          const float* __restrict__ st, const float* __restrict__ g,
                          const float* __restrict__ b, float inv_cnt,
                          float* __restrict__ outE) {
    int gid = blockIdx.x * 256 + threadIdx.x;          // E*128
    int e = gid >> 7, c = gid & 127;
    float v;
    if (c < 64) v = ev[(size_t)e * 64 + c];
    else {
        int ch = c - 64; float sc, sh; bn_coef(st, g, b, ch, inv_cnt, sc, sh);
        v = fmaxf(r1[(size_t)e * 64 + ch] * sc + sh, 0.f);
    }
    outE[gid] = v;
}

__global__ void agg_k(const float* __restrict__ r1, const float* __restrict__ st,
                      const float* __restrict__ g, const float* __restrict__ b,
                      float inv_cnt, float* __restrict__ agg) {
    int gid = blockIdx.x * 256 + threadIdx.x;          // NN*128
    int n = gid >> 7, c = gid & 127;
    float sc, sh; bn_coef(st, g, b, c, inv_cnt, sc, sh);
    float a = 0.f;
    for (int k = 0; k < KN; ++k)
        a += fmaxf(r1[((size_t)n * KN + k) * 128 + c] * sc + sh, 0.f);
    agg[gid] = a * 0.22360679774997896f;               // 1/sqrt(20)
}

__global__ void final_k(const float* __restrict__ x, const float* __restrict__ r1,
                        const float* __restrict__ st, const float* __restrict__ g,
                        const float* __restrict__ b, float inv_cnt,
                        float* __restrict__ out) {
    int gid = blockIdx.x * 256 + threadIdx.x;          // NN*256
    int n = gid >> 8, c = gid & 255;
    float v;
    if (c < 128) v = x[(size_t)n * 128 + c];
    else {
        int ch = c - 128; float sc, sh; bn_coef(st, g, b, ch, inv_cnt, sc, sh);
        v = fmaxf(r1[(size_t)n * 128 + ch] * sc + sh, 0.f);
    }
    out[gid] = v;
}

// --------------------------------------------------------------- launch ----
extern "C" void kernel_launch(void* const* d_in, const int* in_sizes, int n_in,
                              void* d_out, int out_size, void* d_ws, size_t ws_size,
                              hipStream_t stream) {
    (void)in_sizes; (void)n_in; (void)out_size; (void)ws_size;
    auto F = [&](int i) { return (const float*)d_in[i]; };
    const float* x  = F(0);
    const float* ev = F(1);
    const int* adj  = (const int*)d_in[2];

    float* out  = (float*)d_out;
    float* outE = out + (size_t)NN * 256;              // edge_out region

    char* ws = (char*)d_ws;
    float* bufA = (float*)(ws + OFF_BUFA);
    float* bufB = (float*)(ws + OFF_BUFB);
    float* agg  = (float*)(ws + OFF_AGG);
    float* s0   = (float*)(ws + OFF_S0);
    float* s1   = (float*)(ws + OFF_S1);
    int*   deg  = (int*)(ws + OFF_DEG);
    float* st   = (float*)(ws + OFF_ST);

    const float iE = 1.f / (float)E, iN = 1.f / (float)NN;
    const int GE = E / 64, GN = NN / 64;

    hipMemsetAsync(deg, 0, NN * sizeof(int), stream);
    deg_k<<<E / 256, 256, 0, stream>>>(adj, deg);
    stats_k <<<128, 256, 0, stream>>>(x, NN, 128, st + ST_X);
    statsw_k<<<128, 256, 0, stream>>>(x, deg, NN, 128, st + ST_TRG);

    // ---- edge MLP: eu_lin + 2 ResBlocks(64) + BN/ReLU ----
    gemm_k<320, 64, 1><<<GE, 128, 0, stream>>>(nullptr, x, ev, adj, F(3), F(4),
        nullptr, nullptr, nullptr, 0.f, 0, nullptr, bufA);                     // h0
    stats_k<<<64, 256, 0, stream>>>(bufA, E, 64, st + ST_H0);
    gemm_k<64, 64, 0><<<GE, 128, 0, stream>>>(bufA, nullptr, nullptr, nullptr,
        F(7), F(8), st + ST_H0, F(5), F(6), iE, 1, nullptr, bufB);             // h1
    stats_k<<<64, 256, 0, stream>>>(bufB, E, 64, st + ST_H1);
    gemm_k<64, 64, 0><<<GE, 128, 0, stream>>>(bufB, nullptr, nullptr, nullptr,
        F(11), F(12), st + ST_H1, F(9), F(10), iE, 1, bufA, bufB);             // r0e
    stats_k<<<64, 256, 0, stream>>>(bufB, E, 64, st + ST_R0E);
    gemm_k<64, 64, 0><<<GE, 128, 0, stream>>>(bufB, nullptr, nullptr, nullptr,
        F(15), F(16), st + ST_R0E, F(13), F(14), iE, 1, nullptr, bufA);        // h3
    stats_k<<<64, 256, 0, stream>>>(bufA, E, 64, st + ST_H3);
    gemm_k<64, 64, 0><<<GE, 128, 0, stream>>>(bufA, nullptr, nullptr, nullptr,
        F(19), F(20), st + ST_H3, F(17), F(18), iE, 1, bufB, bufB);            // r1e
    stats_k<<<64, 256, 0, stream>>>(bufB, E, 64, st + ST_R1E);
    edgeout_k<<<E * 128 / 256, 256, 0, stream>>>(ev, bufB, st + ST_R1E, F(21),
        F(22), iE, outE);                                                      // edge_out
    stats_k<<<128, 256, 0, stream>>>(outE, E, 128, st + ST_EOUT);
    assemble_k<<<1, 384, 0, stream>>>(st);

    // ---- node encoding: enc_bn0 + enc_lin + 2 ResBlocks(128) ----
    gemm_k<384, 128, 2><<<GE, 128, 0, stream>>>(nullptr, x, outE, adj, F(25),
        F(26), st + ST_NEN, F(23), F(24), iE, 0, nullptr, bufA);               // e0
    stats_k<<<128, 256, 0, stream>>>(bufA, E, 128, st + ST_E0);
    gemm_k<128, 128, 0><<<GE, 128, 0, stream>>>(bufA, nullptr, nullptr, nullptr,
        F(29), F(30), st + ST_E0, F(27), F(28), iE, 1, nullptr, bufB);         // e1
    stats_k<<<128, 256, 0, stream>>>(bufB, E, 128, st + ST_E1);
    gemm_k<128, 128, 0><<<GE, 128, 0, stream>>>(bufB, nullptr, nullptr, nullptr,
        F(33), F(34), st + ST_E1, F(31), F(32), iE, 1, bufA, bufB);            // r0n
    stats_k<<<128, 256, 0, stream>>>(bufB, E, 128, st + ST_R0N);
    gemm_k<128, 128, 0><<<GE, 128, 0, stream>>>(bufB, nullptr, nullptr, nullptr,
        F(37), F(38), st + ST_R0N, F(35), F(36), iE, 1, nullptr, bufA);        // e3
    stats_k<<<128, 256, 0, stream>>>(bufA, E, 128, st + ST_E3);
    gemm_k<128, 128, 0><<<GE, 128, 0, stream>>>(bufA, nullptr, nullptr, nullptr,
        F(41), F(42), st + ST_E3, F(39), F(40), iE, 1, bufB, bufB);            // r1n
    stats_k<<<128, 256, 0, stream>>>(bufB, E, 128, st + ST_R1N);

    // ---- aggregate over K neighbors (after enc_bn1 + relu) ----
    agg_k<<<NN * 128 / 256, 256, 0, stream>>>(bufB, st + ST_R1N, F(43), F(44),
        iE, agg);
    stats_k<<<128, 256, 0, stream>>>(agg, NN, 128, st + ST_AGG);

    // ---- node residual MLP: 2 ResBlocks(128) + res_bn ----
    gemm_k<128, 128, 0><<<GN, 128, 0, stream>>>(agg, nullptr, nullptr, nullptr,
        F(47), F(48), st + ST_AGG, F(45), F(46), iN, 1, nullptr, s0);          // g1
    stats_k<<<128, 256, 0, stream>>>(s0, NN, 128, st + ST_G1);
    gemm_k<128, 128, 0><<<GN, 128, 0, stream>>>(s0, nullptr, nullptr, nullptr,
        F(51), F(52), st + ST_G1, F(49), F(50), iN, 1, agg, s1);               // r0s
    stats_k<<<128, 256, 0, stream>>>(s1, NN, 128, st + ST_R0S);
    gemm_k<128, 128, 0><<<GN, 128, 0, stream>>>(s1, nullptr, nullptr, nullptr,
        F(55), F(56), st + ST_R0S, F(53), F(54), iN, 1, nullptr, s0);          // g3
    stats_k<<<128, 256, 0, stream>>>(s0, NN, 128, st + ST_G3);
    gemm_k<128, 128, 0><<<GN, 128, 0, stream>>>(s0, nullptr, nullptr, nullptr,
        F(59), F(60), st + ST_G3, F(57), F(58), iN, 1, s1, s1);                // r1s
    stats_k<<<128, 256, 0, stream>>>(s1, NN, 128, st + ST_R1S);

    final_k<<<NN * 256 / 256, 256, 0, stream>>>(x, s1, st + ST_R1S, F(61),
        F(62), iN, out);
}
